// PatchGraphEncoder_43731357008476
// MI455X (gfx1250) — compile-verified, matrix operand order checked by reading, not run on hardware
//
#include <hip/hip_runtime.h>

// ---------------------------------------------------------------------------
// PatchGraphEncoder for MI455X (gfx1250): bf16 WMMA everywhere, fused head.
// B=32, N=1024, E=16384, IN=768, HID=512, OUT=400, H=2, F=200, PAIR=800.
// ---------------------------------------------------------------------------

typedef __bf16 v16bf __attribute__((ext_vector_type(16)));
typedef float  v8f   __attribute__((ext_vector_type(8)));
typedef int    v4i   __attribute__((ext_vector_type(4)));

#define WMMA_BF16(a, b, c) \
  __builtin_amdgcn_wmma_f32_16x16x32_bf16(false, (a), false, (b), (short)0, (c), false, false)

#define V8F_ZERO {0.f, 0.f, 0.f, 0.f, 0.f, 0.f, 0.f, 0.f}

static constexpr int B_    = 32;
static constexpr int N_    = 1024;
static constexpr int E_    = 16384;
static constexpr int ROWS  = B_ * N_;        // 32768
static constexpr int ET    = E_ + N_;        // 17408 (with self loops)
static constexpr int RT    = ROWS / 16;      // 2048 row tiles

// fp32 -> bf16 round-to-nearest-even
__device__ __forceinline__ __bf16 f2bf(float f) {
  unsigned int u = __float_as_uint(f);
  unsigned int r = (u + 0x7FFFu + ((u >> 16) & 1u)) >> 16;
  unsigned short s = (unsigned short)r;
  return __builtin_bit_cast(__bf16, s);
}

__device__ __forceinline__ float gelu_f(float x) {
  float x3 = x * x * x;
  return 0.5f * x * (1.0f + tanhf(0.7978845608f * (x + 0.044715f * x3)));
}

// WMMA 16-bit A/B fragment mapping (ISA 7.12.2):
// lane l, element i  ->  (m = l&15, k = ((i>>3)<<4) + ((l>>4)<<3) + (i&7))
__device__ __forceinline__ int frag_k(int lane, int i) {
  return ((i >> 3) << 4) + ((lane >> 4) << 3) + (i & 7);
}
// inverse: (m, k%32) -> linear index inside a 512-element K-slab (lane*16 + i)
__device__ __forceinline__ int slab_index(int m, int kk) {
  int kh = kk >> 4;          // high-K half -> element bit 3
  int rem = kk & 15;
  int lh = rem >> 3;         // lane half
  int kw = rem & 7;
  return ((lh << 4) | m) * 16 + ((kh << 3) | kw);
}

// 16B global -> LDS copy; async path (ASYNCcnt) when available.
// AS1 pointers share the generic 64-bit value; AS3 is 32-bit and the ISA maps
// flat->LDS as addr[31:0], so integer casts give the hardware-correct values.
__device__ __forceinline__ void cp16_g2l(void* lds_dst, const void* gsrc) {
#if __has_builtin(__builtin_amdgcn_global_load_async_to_lds_b128)
  __builtin_amdgcn_global_load_async_to_lds_b128(
      (__attribute__((address_space(1))) v4i*)(unsigned long long)gsrc,
      (__attribute__((address_space(3))) v4i*)(unsigned int)(unsigned long long)lds_dst,
      /*offset*/0, /*cpol*/0);
#else
  *(uint4*)lds_dst = *(const uint4*)gsrc;
#endif
}
__device__ __forceinline__ void async_copy_wait() {
#if __has_builtin(__builtin_amdgcn_global_load_async_to_lds_b128)
#if __has_builtin(__builtin_amdgcn_s_wait_asynccnt)
  __builtin_amdgcn_s_wait_asynccnt(0);
#else
  asm volatile("s_wait_asynccnt 0x0" ::: "memory");
#endif
#endif
}

__device__ __forceinline__ void atomicMaxF(float* addr, float val) {
  unsigned int* ai = (unsigned int*)addr;
  unsigned int old = *ai;
  while (__uint_as_float(old) < val) {
    unsigned int assumed = old;
    old = atomicCAS(ai, assumed, __float_as_uint(val));
    if (old == assumed) break;
  }
}

// ---------------------------------------------------------------------------
// Pack fp32 weight W[K][N] (K-major) into fragment-order bf16 tiles:
// P[nt][kslab][lane][16], zero padded where k>=K or n>=N.
// ---------------------------------------------------------------------------
__global__ void pack_weight_bf16(const float* __restrict__ W, __bf16* __restrict__ P,
                                 int K, int N, int KT, int NT) {
  long long idx = (long long)blockIdx.x * blockDim.x + threadIdx.x;
  long long total = (long long)NT * KT * 512;
  if (idx >= total) return;
  int i    = (int)(idx & 15);
  int lane = (int)((idx >> 4) & 31);
  int slab = (int)((idx >> 9) % KT);
  int nt   = (int)(idx / (512LL * KT));
  int n = nt * 16 + (lane & 15);
  int k = slab * 32 + frag_k(lane, i);
  float v = (k < K && n < N) ? W[(long long)k * N + n] : 0.0f;
  P[idx] = f2bf(v);
}

// ---------------------------------------------------------------------------
// Encoder layer 1: X[32768,768] fp32 -> GELU(X W1 + b1) -> packed bf16 tiles
// ---------------------------------------------------------------------------
__global__ __launch_bounds__(256) void enc1_kernel(
    const float* __restrict__ X, const __bf16* __restrict__ W1p,
    const float* __restrict__ b1, __bf16* __restrict__ Hp) {
  __shared__ alignas(32) __bf16 As[24 * 512];
  const int tid = threadIdx.x;
  const int rt = blockIdx.x;
  const long long rowbase = (long long)rt * 16;
  for (int idx = tid; idx < 16 * 768; idx += 256) {
    int r = idx / 768, k = idx % 768;
    As[(k >> 5) * 512 + slab_index(r, k & 31)] = f2bf(X[(rowbase + r) * 768 + k]);
  }
  __syncthreads();
  const int wave = tid >> 5, lane = tid & 31;
  const int mrow = (lane >> 4) << 3, col = lane & 15;
  for (int nt = wave; nt < 32; nt += 8) {
    v8f acc = V8F_ZERO;
    const __bf16* bp = W1p + (long long)nt * 24 * 512 + lane * 16;
    const __bf16* ap = As + lane * 16;
    for (int s = 0; s < 24; ++s) {
      v16bf a = *(const v16bf*)(ap + s * 512);
      v16bf b = *(const v16bf*)(bp + s * 512);
      acc = WMMA_BF16(a, b, acc);
    }
    int gcol = nt * 16 + col;
    __bf16* hp = Hp + (long long)rt * (16 * 512);
#pragma unroll
    for (int d = 0; d < 8; ++d) {
      int row = mrow + d;
      float g = gelu_f(acc[d] + b1[gcol]);
      hp[(gcol >> 5) * 512 + slab_index(row, gcol & 31)] = f2bf(g);
    }
  }
}

// ---------------------------------------------------------------------------
// Encoder layer 2: Hp (packed) x W2 + b2 -> enc.
// Writes: feat (d_out) = enc + gat_b   and packed bf16 enc (Kp=416, pad pre-zeroed)
// ---------------------------------------------------------------------------
__global__ __launch_bounds__(256) void enc2_kernel(
    const __bf16* __restrict__ Hp, const __bf16* __restrict__ W2p,
    const float* __restrict__ b2, const float* __restrict__ gatb,
    float* __restrict__ feat, __bf16* __restrict__ Ep) {
  const int tid = threadIdx.x, rt = blockIdx.x;
  const int wave = tid >> 5, lane = tid & 31;
  const int mrow = (lane >> 4) << 3, col = lane & 15;
  const __bf16* ap = Hp + (long long)rt * (16 * 512) + lane * 16;
  for (int nt = wave; nt < 25; nt += 8) {
    v8f acc = V8F_ZERO;
    const __bf16* bp = W2p + (long long)nt * 16 * 512 + lane * 16;
    for (int s = 0; s < 16; ++s) {
      v16bf a = *(const v16bf*)(ap + s * 512);
      v16bf b = *(const v16bf*)(bp + s * 512);
      acc = WMMA_BF16(a, b, acc);
    }
    int gcol = nt * 16 + col;
#pragma unroll
    for (int d = 0; d < 8; ++d) {
      int row = mrow + d;
      long long grow = (long long)rt * 16 + row;
      float e = acc[d] + b2[gcol];
      feat[grow * 400 + gcol] = e + gatb[gcol];
      Ep[(long long)rt * (13 * 512) + (gcol >> 5) * 512 + slab_index(row, gcol & 31)] = f2bf(e);
    }
  }
}

// ---------------------------------------------------------------------------
// GAT linear: wh = enc @ gat_w  (Kp=416 padded) -> wh[32768,400] fp32
// ---------------------------------------------------------------------------
__global__ __launch_bounds__(256) void wh_kernel(
    const __bf16* __restrict__ Ep, const __bf16* __restrict__ GWp,
    float* __restrict__ wh) {
  const int tid = threadIdx.x, rt = blockIdx.x;
  const int wave = tid >> 5, lane = tid & 31;
  const int mrow = (lane >> 4) << 3, col = lane & 15;
  const __bf16* ap = Ep + (long long)rt * (13 * 512) + lane * 16;
  for (int nt = wave; nt < 25; nt += 8) {
    v8f acc = V8F_ZERO;
    const __bf16* bp = GWp + (long long)nt * 13 * 512 + lane * 16;
    for (int s = 0; s < 13; ++s) {
      v16bf a = *(const v16bf*)(ap + s * 512);
      v16bf b = *(const v16bf*)(bp + s * 512);
      acc = WMMA_BF16(a, b, acc);
    }
    int gcol = nt * 16 + col;
#pragma unroll
    for (int d = 0; d < 8; ++d) {
      long long grow = (long long)rt * 16 + mrow + d;
      wh[grow * 400 + gcol] = acc[d];
    }
  }
}

// a_s / a_d attention projections
__global__ void att_kernel(const float* __restrict__ wh, const float* __restrict__ att_src,
                           const float* __restrict__ att_dst,
                           float* __restrict__ as_, float* __restrict__ ad_) {
  int idx = blockIdx.x * blockDim.x + threadIdx.x;
  if (idx >= ROWS * 2) return;
  int row = idx >> 1, h = idx & 1;
  const float* w = wh + (long long)row * 400 + h * 200;
  const float* s = att_src + h * 200;
  const float* d = att_dst + h * 200;
  float ss = 0.f, dd = 0.f;
  for (int f = 0; f < 200; ++f) { float v = w[f]; ss += v * s[f]; dd += v * d[f]; }
  as_[row * 2 + h] = ss;
  ad_[row * 2 + h] = dd;
}

__global__ void init_mz_kernel(float* __restrict__ m, float* __restrict__ z) {
  int idx = blockIdx.x * blockDim.x + threadIdx.x;
  if (idx >= ROWS * 2) return;
  m[idx] = -3.0e38f;
  z[idx] = 0.f;
}

// e = leaky(a_s[src]+a_d[dst]); segment max into m[dst]
__global__ void edge_logits_kernel(const int* __restrict__ edges,
                                   const float* __restrict__ as_, const float* __restrict__ ad_,
                                   float* __restrict__ ebuf, float* __restrict__ m) {
  int idx = blockIdx.x * blockDim.x + threadIdx.x;
  if (idx >= B_ * ET) return;
  int et = idx % ET, b = idx / ET;
  int s, d;
  if (et < E_) { s = edges[et]; d = edges[E_ + et]; } else { s = d = et - E_; }
  int rs = b * N_ + s, rd = b * N_ + d;
#pragma unroll
  for (int h = 0; h < 2; ++h) {
    float e = as_[rs * 2 + h] + ad_[rd * 2 + h];
    e = e > 0.f ? e : 0.2f * e;
    ebuf[(long long)idx * 2 + h] = e;
    atomicMaxF(&m[rd * 2 + h], e);
  }
}

// ee = exp(e - m[dst]); segment sum into z[dst]
__global__ void edge_exp_kernel(const int* __restrict__ edges, const float* __restrict__ m,
                                float* __restrict__ ebuf, float* __restrict__ z) {
  int idx = blockIdx.x * blockDim.x + threadIdx.x;
  if (idx >= B_ * ET * 2) return;
  int h = idx & 1;
  int et = (idx >> 1) % ET, b = (idx >> 1) / ET;
  int d = (et < E_) ? edges[E_ + et] : et - E_;
  int rd = b * N_ + d;
  float ee = __expf(ebuf[idx] - m[rd * 2 + h]);
  ebuf[idx] = ee;
  atomicAdd(&z[rd * 2 + h], ee);
}

// feat[dst] += (ee/z[dst]) * wh[src]  (residual: feat already holds enc+gat_b)
__global__ void edge_agg_kernel(const int* __restrict__ edges, const float* __restrict__ ebuf,
                                const float* __restrict__ z, const float* __restrict__ wh,
                                float* __restrict__ feat) {
  long long idx = (long long)blockIdx.x * blockDim.x + threadIdx.x;
  if (idx >= (long long)B_ * ET * 400) return;
  int col = (int)(idx % 400);
  long long r = idx / 400;
  int et = (int)(r % ET), b = (int)(r / ET);
  int s, d;
  if (et < E_) { s = edges[et]; d = edges[E_ + et]; } else { s = d = et - E_; }
  int h = col >= 200;
  int rd = b * N_ + d, rs = b * N_ + s;
  float coef = ebuf[((long long)b * ET + et) * 2 + h] / z[rd * 2 + h];
  atomicAdd(&feat[(long long)rd * 400 + col], coef * wh[(long long)rs * 400 + col]);
}

__global__ void feat2bf_kernel(const float* __restrict__ feat, __bf16* __restrict__ fb) {
  long long idx = (long long)blockIdx.x * blockDim.x + threadIdx.x;
  if (idx >= (long long)ROWS * 400) return;
  fb[idx] = f2bf(feat[idx]);
}

// ---------------------------------------------------------------------------
// Fused predict head: one workgroup = 32 edges of one batch (two 16-row panels,
// dual accumulators per wave -> each B fragment feeds 2 WMMAs, halving L2 reads).
// L1 800->800 GELU, L2 800->400 GELU, L3 400->3, all in LDS via WMMA.
// A-panel gather uses async global->LDS 16B copies (ASYNCcnt) when available.
// ---------------------------------------------------------------------------
__global__ __launch_bounds__(256) void head_kernel(
    const int* __restrict__ edges, const __bf16* __restrict__ fb,
    const __bf16* __restrict__ W1p, const float* __restrict__ hb1,
    const __bf16* __restrict__ W2p, const float* __restrict__ hb2,
    const __bf16* __restrict__ W3p, const float* __restrict__ hb3,
    float* __restrict__ pred) {
  __shared__ alignas(32) __bf16 A1[2][25 * 512];
  __shared__ alignas(32) __bf16 A2[2][25 * 512];
  __shared__ alignas(32) __bf16 A3[2][13 * 512];
  const int tid = threadIdx.x;
  const int b = blockIdx.x >> 9;       // batch (512 tiles of 32 edges each)
  const int t = blockIdx.x & 511;

  // gather pair rows [32 x 800] = feat[src] || feat[dst], 16B chunks (8 bf16)
  for (int c = tid; c < 32 * 100; c += 256) {
    int r = c / 100;                   // 0..31
    int k0 = (c % 100) * 8;            // 0..792
    int e = t * 32 + r;
    int node, kk0;
    if (k0 < 400) { node = edges[e]; kk0 = k0; }
    else          { node = edges[E_ + e]; kk0 = k0 - 400; }
    const __bf16* src = fb + (long long)(b * N_ + node) * 400 + kk0;
    int half = r >> 4, rr = r & 15;
    int slab = k0 >> 5, kk = k0 & 31;
    int kh = kk >> 4, lh = (kk >> 3) & 1;
    cp16_g2l(&A1[half][slab * 512 + (lh * 16 + rr) * 16 + kh * 8], src);
  }
  // zero K-pad of A3 (k = 400..415), 16B chunks
  for (int c = tid; c < 2 * 13 * 512 * 2 / 16; c += 256) {
    ((uint4*)A3)[c] = uint4{0u, 0u, 0u, 0u};
  }
  async_copy_wait();
  __syncthreads();

  const int wave = tid >> 5, lane = tid & 31;
  const int mrow = (lane >> 4) << 3, col = lane & 15;

  // layer 1: 50 column tiles, K = 25 slabs, 2 row panels share each B fragment
  for (int nt = wave; nt < 50; nt += 8) {
    v8f acc0 = V8F_ZERO, acc1 = V8F_ZERO;
    const __bf16* bp = W1p + (long long)nt * 25 * 512 + lane * 16;
    for (int s = 0; s < 25; ++s) {
      v16bf w = *(const v16bf*)(bp + s * 512);
      v16bf a0 = *(const v16bf*)(&A1[0][s * 512 + lane * 16]);
      v16bf a1 = *(const v16bf*)(&A1[1][s * 512 + lane * 16]);
      acc0 = WMMA_BF16(a0, w, acc0);
      acc1 = WMMA_BF16(a1, w, acc1);
    }
    int gcol = nt * 16 + col;
    float bias = hb1[gcol];
#pragma unroll
    for (int d = 0; d < 8; ++d) {
      int row = mrow + d;
      int si = (gcol >> 5) * 512 + slab_index(row, gcol & 31);
      A2[0][si] = f2bf(gelu_f(acc0[d] + bias));
      A2[1][si] = f2bf(gelu_f(acc1[d] + bias));
    }
  }
  __syncthreads();

  // layer 2: 25 column tiles, K = 25 slabs
  for (int nt = wave; nt < 25; nt += 8) {
    v8f acc0 = V8F_ZERO, acc1 = V8F_ZERO;
    const __bf16* bp = W2p + (long long)nt * 25 * 512 + lane * 16;
    for (int s = 0; s < 25; ++s) {
      v16bf w = *(const v16bf*)(bp + s * 512);
      v16bf a0 = *(const v16bf*)(&A2[0][s * 512 + lane * 16]);
      v16bf a1 = *(const v16bf*)(&A2[1][s * 512 + lane * 16]);
      acc0 = WMMA_BF16(a0, w, acc0);
      acc1 = WMMA_BF16(a1, w, acc1);
    }
    int gcol = nt * 16 + col;
    float bias = hb2[gcol];
#pragma unroll
    for (int d = 0; d < 8; ++d) {
      int row = mrow + d;
      int si = (gcol >> 5) * 512 + slab_index(row, gcol & 31);
      A3[0][si] = f2bf(gelu_f(acc0[d] + bias));
      A3[1][si] = f2bf(gelu_f(acc1[d] + bias));
    }
  }
  __syncthreads();

  // layer 3: single 16-col tile (3 valid), K = 13 slabs (padded); waves 0,1
  if (wave < 2) {
    v8f acc = V8F_ZERO;
    const __bf16* bp = W3p + lane * 16;
    for (int s = 0; s < 13; ++s) {
      v16bf a = *(const v16bf*)(&A3[wave][s * 512 + lane * 16]);
      v16bf w = *(const v16bf*)(bp + s * 512);
      acc = WMMA_BF16(a, w, acc);
    }
    if (col < 3) {
#pragma unroll
      for (int d = 0; d < 8; ++d) {
        long long e = (long long)b * E_ + t * 32 + wave * 16 + mrow + d;
        pred[e * 3 + col] = acc[d] + hb3[col];
      }
    }
  }
}

// ---------------------------------------------------------------------------
// Host-side orchestration
// ---------------------------------------------------------------------------
static inline char* align256(char* p) {
  return (char*)(((uintptr_t)p + 255u) & ~(uintptr_t)255u);
}

extern "C" void kernel_launch(void* const* d_in, const int* in_sizes, int n_in,
                              void* d_out, int out_size, void* d_ws, size_t ws_size,
                              hipStream_t stream) {
  (void)in_sizes; (void)n_in; (void)out_size; (void)ws_size;
  const float* X       = (const float*)d_in[0];
  const int*   edges   = (const int*)d_in[1];
  const float* enc_w1  = (const float*)d_in[2];
  const float* enc_b1  = (const float*)d_in[3];
  const float* enc_w2  = (const float*)d_in[4];
  const float* enc_b2  = (const float*)d_in[5];
  const float* gat_w   = (const float*)d_in[6];   // [400][2][200] == [400][400]
  const float* att_src = (const float*)d_in[7];
  const float* att_dst = (const float*)d_in[8];
  const float* gat_b   = (const float*)d_in[9];
  const float* h_w1    = (const float*)d_in[10];
  const float* h_b1    = (const float*)d_in[11];
  const float* h_w2    = (const float*)d_in[12];
  const float* h_b2    = (const float*)d_in[13];
  const float* h_w3    = (const float*)d_in[14];
  const float* h_b3    = (const float*)d_in[15];

  float* feat = (float*)d_out;                       // [32768,400]
  float* pred = feat + (long long)ROWS * 400;        // [B*E,3]

  // workspace carving
  char* p = (char*)d_ws;
  auto take_bf = [&](long long n) { __bf16* r = (__bf16*)p; p = align256(p + n * 2); return r; };
  auto take_f  = [&](long long n) { float*  r = (float*)p;  p = align256(p + n * 4); return r; };

  __bf16* W1p  = take_bf(32LL * 24 * 512);   // enc_w1 packed
  __bf16* W2p  = take_bf(25LL * 16 * 512);   // enc_w2 packed
  __bf16* GWp  = take_bf(25LL * 13 * 512);   // gat_w packed (Kp=416)
  __bf16* HW1p = take_bf(50LL * 25 * 512);   // h_w1 packed
  __bf16* HW2p = take_bf(25LL * 25 * 512);   // h_w2 packed
  __bf16* HW3p = take_bf(1LL * 13 * 512);    // h_w3 packed (Np=16, Kp=416)
  __bf16* Hp   = take_bf((long long)RT * 16 * 512);  // hidden packed
  __bf16* Ep   = take_bf((long long)RT * 13 * 512);  // enc packed (Kp=416)
  __bf16* FB   = take_bf((long long)ROWS * 400);     // feat bf16 row-major
  float*  WH   = take_f((long long)ROWS * 400);
  float*  ASb  = take_f(ROWS * 2);
  float*  ADb  = take_f(ROWS * 2);
  float*  Mb   = take_f(ROWS * 2);
  float*  Zb   = take_f(ROWS * 2);
  float*  EB   = take_f((long long)B_ * ET * 2);

  const int TB = 256;
  auto blocks = [](long long n, int tb) { return (unsigned)((n + tb - 1) / tb); };

  // 1. pack all weights to fragment-order bf16
  pack_weight_bf16<<<blocks(32LL * 24 * 512, TB), TB, 0, stream>>>(enc_w1, W1p, 768, 512, 24, 32);
  pack_weight_bf16<<<blocks(25LL * 16 * 512, TB), TB, 0, stream>>>(enc_w2, W2p, 512, 400, 16, 25);
  pack_weight_bf16<<<blocks(25LL * 13 * 512, TB), TB, 0, stream>>>(gat_w,  GWp, 400, 400, 13, 25);
  pack_weight_bf16<<<blocks(50LL * 25 * 512, TB), TB, 0, stream>>>(h_w1,  HW1p, 800, 800, 25, 50);
  pack_weight_bf16<<<blocks(25LL * 25 * 512, TB), TB, 0, stream>>>(h_w2,  HW2p, 800, 400, 25, 25);
  pack_weight_bf16<<<blocks(1LL  * 13 * 512, TB), TB, 0, stream>>>(h_w3,  HW3p, 400, 3,  13, 1);

  // 2. encoder
  enc1_kernel<<<RT, TB, 0, stream>>>(X, W1p, enc_b1, Hp);
  (void)hipMemsetAsync(Ep, 0, (size_t)RT * 13 * 512 * 2, stream);  // zero K-pad slab
  enc2_kernel<<<RT, TB, 0, stream>>>(Hp, W2p, enc_b2, gat_b, feat, Ep);

  // 3. GAT
  wh_kernel<<<RT, TB, 0, stream>>>(Ep, GWp, WH);
  att_kernel<<<blocks(ROWS * 2, TB), TB, 0, stream>>>(WH, att_src, att_dst, ASb, ADb);
  init_mz_kernel<<<blocks(ROWS * 2, TB), TB, 0, stream>>>(Mb, Zb);
  edge_logits_kernel<<<blocks((long long)B_ * ET, TB), TB, 0, stream>>>(edges, ASb, ADb, EB, Mb);
  edge_exp_kernel<<<blocks((long long)B_ * ET * 2, TB), TB, 0, stream>>>(edges, Mb, EB, Zb);
  edge_agg_kernel<<<blocks((long long)B_ * ET * 400, TB), TB, 0, stream>>>(edges, EB, Zb, WH, feat);

  // 4. fused predict head (32 edges / workgroup)
  feat2bf_kernel<<<blocks((long long)ROWS * 400, TB), TB, 0, stream>>>(feat, FB);
  head_kernel<<<B_ * (E_ / 32), TB, 0, stream>>>(edges, FB, HW1p, h_b1, HW2p, h_b2, HW3p, h_b3, pred);
}
